// SelfAttention_30185030157056
// MI455X (gfx1250) — matrix operand order repaired and checked by reading, not compiled, not run on hardware
//
#include <hip/hip_runtime.h>

#define B_ 4
#define S_ 2048
#define E_ 1024
#define H_ 16
#define D_ 64

typedef __attribute__((ext_vector_type(16))) _Float16 v16h;
typedef __attribute__((ext_vector_type(8)))  _Float16 v8h;
typedef __attribute__((ext_vector_type(8)))  float    v8f;

static __device__ __forceinline__ v8f wmma_f16(v16h a, v16h b, v8f c) {
  // D = A(16x32 f16) * B(32x16 f16) + C(16x16 f32)
  return __builtin_amdgcn_wmma_f32_16x16x32_f16(false, a, false, b, (short)0, c,
                                                false, false);
}

// A-fragment (16x32 f16): lane holds row M=l16, K = hf*16 .. hf*16+15
static __device__ __forceinline__ v16h ld_fragA(const _Float16* p, int ld,
                                                int row0, int k0, int l16, int hf) {
  const _Float16* q = p + (size_t)(row0 + l16) * ld + k0 + hf * 16;
  v8h lo = *(const v8h*)(q);
  v8h hi = *(const v8h*)(q + 8);
  v16h r;
#pragma unroll
  for (int i = 0; i < 8; ++i) { r[i] = lo[i]; r[i + 8] = hi[i]; }
  return r;
}

// B-fragment (32x16 f16): B[k][n] = src[n][k], src row-major [N x ld].
// lane holds col N=l16, K = hf*16 .. hf*16+15 (one contiguous 32B read).
static __device__ __forceinline__ v16h ld_fragB(const _Float16* p, int ld,
                                                int n0, int k0, int l16, int hf) {
  const _Float16* q = p + (size_t)(n0 + l16) * ld + k0 + hf * 16;
  return *(const v16h*)(q);
}

// ---- CDNA5 async copy: global -> LDS, 16B per lane, tracked by ASYNCcnt
static __device__ __forceinline__ void async_load_b128(void* lds_ptr, const void* gptr) {
  // low 32 bits of a generic LDS address are the DS byte offset
  unsigned lds_off = (unsigned)(unsigned long long)lds_ptr;
  unsigned long long ga = (unsigned long long)gptr;
  asm volatile("global_load_async_to_lds_b128 %0, %1, off"
               :: "v"(lds_off), "v"(ga) : "memory");
}
static __device__ __forceinline__ void wait_async0() {
  asm volatile("s_wait_asynccnt 0" ::: "memory");
}

// ---------------------------------------------------------------- convert
__global__ __launch_bounds__(256) void f32_to_f16(const float* __restrict__ in,
                                                  _Float16* __restrict__ out, int n) {
  int i = blockIdx.x * 256 + threadIdx.x;
  if (i < n) out[i] = (_Float16)in[i];
}

// ------------------------------------------------------- QKV projection GEMM
__global__ __launch_bounds__(256) void qkv_proj(const _Float16* __restrict__ Xh,
                                                const _Float16* __restrict__ Wh,
                                                const float* __restrict__ bias,
                                                _Float16* __restrict__ Out,
                                                int transposed) {
  const int tid = threadIdx.x;
  const int lane = tid & 31, wid = tid >> 5;
  const int l16 = lane & 15, hf = lane >> 4;
  const int wm = wid & 1, wn = wid >> 1;
  const int m_base = blockIdx.y * 64 + wm * 32;
  const int n_base = blockIdx.x * 128 + wn * 32;

  v8f acc[2][2] = {};
  for (int k0 = 0; k0 < E_; k0 += 32) {
    v16h a0 = ld_fragA(Xh, E_, m_base,      k0, l16, hf);
    v16h a1 = ld_fragA(Xh, E_, m_base + 16, k0, l16, hf);
    v16h b0 = ld_fragB(Wh, E_, n_base,      k0, l16, hf);
    v16h b1 = ld_fragB(Wh, E_, n_base + 16, k0, l16, hf);
    acc[0][0] = wmma_f16(a0, b0, acc[0][0]);
    acc[0][1] = wmma_f16(a0, b1, acc[0][1]);
    acc[1][0] = wmma_f16(a1, b0, acc[1][0]);
    acc[1][1] = wmma_f16(a1, b1, acc[1][1]);
  }

  float bias_n[2];
#pragma unroll
  for (int j = 0; j < 2; ++j) bias_n[j] = bias[n_base + 16 * j + l16];

#pragma unroll
  for (int i = 0; i < 2; ++i)
#pragma unroll
    for (int j = 0; j < 2; ++j)
#pragma unroll
      for (int r = 0; r < 8; ++r) {
        int m = m_base + 16 * i + r + 8 * hf;      // flat (b,s)
        int n = n_base + 16 * j + l16;             // flat (h,d)
        float val = acc[i][j][r] + bias_n[j];
        int bb = m >> 11, s = m & (S_ - 1);
        int hh = n >> 6,  d = n & (D_ - 1);
        size_t off = transposed
            ? ((((size_t)bb * H_ + hh) * D_ + d) * S_ + s)   // V^T: [B,H,D,S]
            : ((((size_t)bb * H_ + hh) * S_ + s) * D_ + d);  // Q,K: [B,H,S,D]
        Out[off] = (_Float16)val;
      }
}

// ------------------------------------------------------------ flash attention
// 8 waves * 16 q-rows = 128 q-rows per block; kv streamed in 64-wide chunks.
// K/V tiles staged cooperatively in LDS via async copies, double-buffered.
static __device__ __forceinline__ void issue_tile_loads(const _Float16* Kg,
                                                        const _Float16* Vg,
                                                        int kv0,
                                                        _Float16* sKbuf,
                                                        _Float16* sVbuf,
                                                        int tid) {
  const int row = tid >> 3;           // 0..31
  const int col = (tid & 7) * 8;      // f16 elements, 16B granules
  // K tile: [kv 64][d 64] row-major
  async_load_b128(sKbuf + row * 64 + col,        Kg + (size_t)(kv0 + row) * D_ + col);
  async_load_b128(sKbuf + (row + 32) * 64 + col, Kg + (size_t)(kv0 + row + 32) * D_ + col);
  // V tile: [d 64][kv 64] row-major (V already transposed in HBM)
  async_load_b128(sVbuf + row * 64 + col,        Vg + (size_t)row * S_ + kv0 + col);
  async_load_b128(sVbuf + (row + 32) * 64 + col, Vg + (size_t)(row + 32) * S_ + kv0 + col);
}

__global__ __launch_bounds__(256) void flash_attn(const _Float16* __restrict__ Q,
                                                  const _Float16* __restrict__ K,
                                                  const _Float16* __restrict__ Vt,
                                                  float* __restrict__ Out) {
  __shared__ _Float16 sK[2][64 * 64];   // 2 x 8 KB
  __shared__ _Float16 sV[2][64 * 64];   // 2 x 8 KB
  __shared__ _Float16 sP[8][16 * 64];   // 16 KB P staging (per-wave private)

  const int tid = threadIdx.x;
  const int lane = tid & 31, wid = tid >> 5;
  const int l16 = lane & 15, hf = lane >> 4;
  const int b = blockIdx.z, h = blockIdx.y;
  const size_t bh = (size_t)b * H_ + h;
  const _Float16* Qg = Q  + bh * S_ * D_;
  const _Float16* Kg = K  + bh * S_ * D_;
  const _Float16* Vg = Vt + bh * (size_t)D_ * S_;
  const int m_base = blockIdx.x * 128 + wid * 16;
  _Float16* Pst = sP[wid];

  // Q fragments for this wave's 16 rows, pre-scaled by 1/sqrt(D)
  v16h qa0 = ld_fragA(Qg, D_, m_base, 0,  l16, hf);
  v16h qa1 = ld_fragA(Qg, D_, m_base, 32, l16, hf);
  qa0 = qa0 * (_Float16)0.125f;
  qa1 = qa1 * (_Float16)0.125f;

  v16h vones;
#pragma unroll
  for (int i = 0; i < 16; ++i) vones[i] = (_Float16)1.0f;

  v8f acc[4] = {};   // O accumulator 16x64
  v8f accl = {};     // running row-sum l via P*ones (ones-trick)
  float m_i[8];
#pragma unroll
  for (int r = 0; r < 8; ++r) m_i[r] = -1e30f;

  issue_tile_loads(Kg, Vg, 0, sK[0], sV[0], tid);
  wait_async0();
  __syncthreads();

  const int NC = S_ / 64;
  for (int c = 0; c < NC; ++c) {
    const int cur = c & 1;
    if (c + 1 < NC)
      issue_tile_loads(Kg, Vg, (c + 1) * 64, sK[1 - cur], sV[1 - cur], tid);

    const _Float16* Kt = sK[cur];
    const _Float16* Vti = sV[cur];

    // ---- scores S = (Q/sqrtD) K^T : 16x64, 4 N-tiles x 2 K-steps
    v8f sc[4] = {};
#pragma unroll
    for (int ks = 0; ks < 2; ++ks) {
      v16h qa = (ks == 0) ? qa0 : qa1;
#pragma unroll
      for (int nt = 0; nt < 4; ++nt) {
        v16h bk = *(const v16h*)(Kt + (nt * 16 + l16) * 64 + ks * 32 + hf * 16);
        sc[nt] = wmma_f16(qa, bk, sc[nt]);
      }
    }

    // ---- online softmax (max reduction only; sums come from P*ones WMMA)
#pragma unroll
    for (int r = 0; r < 8; ++r) {
      float mx = fmaxf(fmaxf(sc[0][r], sc[1][r]), fmaxf(sc[2][r], sc[3][r]));
#pragma unroll
      for (int mk = 8; mk >= 1; mk >>= 1) mx = fmaxf(mx, __shfl_xor(mx, mk, 32));
      float m_new = fmaxf(m_i[r], mx);
      float corr = __expf(m_i[r] - m_new);
      m_i[r] = m_new;
      const int row = (r + 8 * hf) * 64;
#pragma unroll
      for (int nt = 0; nt < 4; ++nt) {
        float p = __expf(sc[nt][r] - m_new);
        acc[nt][r] *= corr;
        Pst[row + nt * 16 + l16] = (_Float16)p;
      }
      accl[r] *= corr;
    }

    // ---- reload P as A-fragments (layout fixup via wave-private LDS)
    v16h pa0, pa1;
#pragma unroll
    for (int ks = 0; ks < 2; ++ks) {
      const _Float16* q = Pst + l16 * 64 + ks * 32 + hf * 16;
      v8h lo = *(const v8h*)(q);
      v8h hi = *(const v8h*)(q + 8);
#pragma unroll
      for (int i = 0; i < 8; ++i) {
        if (ks == 0) { pa0[i] = lo[i]; pa0[i + 8] = hi[i]; }
        else         { pa1[i] = lo[i]; pa1[i + 8] = hi[i]; }
      }
    }

    // ---- O += P*V ; l += P*1
#pragma unroll
    for (int ks = 0; ks < 2; ++ks) {
      v16h pa = (ks == 0) ? pa0 : pa1;
#pragma unroll
      for (int nt = 0; nt < 4; ++nt) {
        v16h bv = *(const v16h*)(Vti + (nt * 16 + l16) * 64 + ks * 32 + hf * 16);
        acc[nt] = wmma_f16(pa, bv, acc[nt]);
      }
      accl = wmma_f16(pa, vones, accl);
    }

    wait_async0();     // next chunk's tiles landed in the other buffer
    __syncthreads();   // everyone done reading current buffers
  }

  // ---- normalize and store fp32 output [B,S,H,D]
#pragma unroll
  for (int r = 0; r < 8; ++r) {
    float inv = 1.f / accl[r];
    int s = m_base + r + 8 * hf;
#pragma unroll
    for (int nt = 0; nt < 4; ++nt) {
      int d = nt * 16 + l16;
      Out[(((size_t)b * S_ + s) * H_ + h) * D_ + d] = acc[nt][r] * inv;
    }
  }
}

// ---------------------------------------------------------------- launcher
extern "C" void kernel_launch(void* const* d_in, const int* in_sizes, int n_in,
                              void* d_out, int out_size, void* d_ws, size_t ws_size,
                              hipStream_t stream) {
  const float* x  = (const float*)d_in[0];
  const float* Wq = (const float*)d_in[1];
  const float* bq = (const float*)d_in[2];
  const float* Wk = (const float*)d_in[3];
  const float* bk = (const float*)d_in[4];
  const float* Wv = (const float*)d_in[5];
  const float* bv = (const float*)d_in[6];
  float* out = (float*)d_out;

  char* ws = (char*)d_ws;
  _Float16* Xh  = (_Float16*)(ws);                        // 16 MB
  _Float16* Whq = (_Float16*)(ws + (16ull << 20));        //  2 MB
  _Float16* Whk = (_Float16*)(ws + (18ull << 20));        //  2 MB
  _Float16* Whv = (_Float16*)(ws + (20ull << 20));        //  2 MB
  _Float16* Qb  = (_Float16*)(ws + (22ull << 20));        // 16 MB  [B,H,S,D]
  _Float16* Kb  = (_Float16*)(ws + (38ull << 20));        // 16 MB  [B,H,S,D]
  _Float16* Vtb = (_Float16*)(ws + (54ull << 20));        // 16 MB  [B,H,D,S]

  const int nx = B_ * S_ * E_;
  const int nw = E_ * E_;
  f32_to_f16<<<(nx + 255) / 256, 256, 0, stream>>>(x,  Xh,  nx);
  f32_to_f16<<<(nw + 255) / 256, 256, 0, stream>>>(Wq, Whq, nw);
  f32_to_f16<<<(nw + 255) / 256, 256, 0, stream>>>(Wk, Whk, nw);
  f32_to_f16<<<(nw + 255) / 256, 256, 0, stream>>>(Wv, Whv, nw);

  dim3 pg(E_ / 128, (B_ * S_) / 64, 1);
  qkv_proj<<<pg, 256, 0, stream>>>(Xh, Whq, bq, Qb,  0);
  qkv_proj<<<pg, 256, 0, stream>>>(Xh, Whk, bk, Kb,  0);
  qkv_proj<<<pg, 256, 0, stream>>>(Xh, Whv, bv, Vtb, 1);

  dim3 ag(S_ / 128, H_, B_);
  flash_attn<<<ag, 256, 0, stream>>>(Qb, Kb, Vtb, out);
}
